// S2S_28089086116414
// MI455X (gfx1250) — matrix operand order repaired and checked by reading, not compiled, not run on hardware
//
#include <hip/hip_runtime.h>

// Seq2seq LSTM (enc->dec) + vocab projection for MI455X (gfx1250, wave32, WMMA).
// B=1024, S=256, EMB=32, HID=64, VOCAB=50.
//
// Strategy:
//  - Precompute per-token gate table G[v][4H] = E[v] @ W_ih^T + b_ih + b_hh
//    (kills the big einsum; xg becomes a 50-entry LUT).
//  - One WG = 16 batch rows, 4 waves. Wave w computes gate slice [64w,64w+64)
//    of h_{t-1} @ W_hh^T via v_wmma_f32_16x16x32_f16 (8 WMMAs/step/wave),
//    with the xg LUT row loaded directly into the WMMA C accumulator.
//  - All activations are branchless amp*sigmoid(scale*x)+off using native
//    v_exp_f32 + v_rcp_f32 (tanh(x) = 2*sigmoid(2x)-1): 3 VALU ops each,
//    no EXEC divergence, no IEEE-division expansion on the serial chain.
//  - c state lives in fp32 registers; h round-trips through LDS as f16.
//  - Encoder and decoder run in the same kernel so (h,c) never hits HBM.
//  - Decoder fuses the [16x64]x[64x50] vocab projection as 2 more WMMAs/wave.

static constexpr int kVOCAB = 50;
static constexpr int kEMB   = 32;
static constexpr int kHID   = 64;
static constexpr int kG4    = 4 * kHID;   // 256
static constexpr int kB     = 1024;
static constexpr int kS     = 256;

// -log2(e), -2*log2(e): sigmoid(x) = rcp(1 + 2^(-x*log2e))
static constexpr float kNL2E  = -1.4426950408889634f;
static constexpr float kN2L2E = -2.8853900817779268f;

typedef __attribute__((ext_vector_type(16))) _Float16 v16h;
typedef __attribute__((ext_vector_type(8)))  float    v8f;

__device__ __forceinline__ float fast_exp2(float x) { return __builtin_amdgcn_exp2f(x); }
__device__ __forceinline__ float fast_rcp(float x)  { return __builtin_amdgcn_rcpf(x); }

// tanh via native exp2/rcp: tanh(x) = 2/(1 + 2^(-2x*log2e)) - 1
__device__ __forceinline__ float fast_tanh(float x) {
    return __builtin_fmaf(2.0f, fast_rcp(1.0f + fast_exp2(kN2L2E * x)), -1.0f);
}

// ---------------------------------------------------------------------------
// Kernel 1: per-token gate-preactivation table
//   G[v][g] = sum_e E[v,e] * Wih[g,e] + bih[g] + bhh[g]
// ---------------------------------------------------------------------------
__global__ void gate_table_kernel(const float* __restrict__ E,
                                  const float* __restrict__ Wih,
                                  const float* __restrict__ bih,
                                  const float* __restrict__ bhh,
                                  float* __restrict__ G)
{
    const int g = threadIdx.x;            // 0..255
    const float bias = bih[g] + bhh[g];
    for (int v = 0; v < kVOCAB; ++v) {
        float acc = bias;
        #pragma unroll
        for (int e = 0; e < kEMB; ++e)
            acc += E[v * kEMB + e] * Wih[g * kEMB + e];
        G[v * kG4 + g] = acc;
    }
}

// ---------------------------------------------------------------------------
// A-fragment (16x32 f16, contiguous-K order, matching the B fragments) of the h tile held in LDS.
// lane 0-15: row M=lane, K = {0..15}+32*kb ; lanes 16-31: K +16.
// Loaded as 8 dword pairs (K pairs are contiguous f16 in the row).
// ---------------------------------------------------------------------------
__device__ __forceinline__ v16h load_hfrag(const _Float16* __restrict__ hbuf,
                                           int kb, int lane)
{
    union { v16h h; unsigned u[8]; } au;
    const int m     = lane & 15;
    const int kbase = kb * 32 + ((lane >> 4) << 4);
    #pragma unroll
    for (int p = 0; p < 8; ++p) {
        const int k = kbase + p * 2;
        au.u[p] = *(const unsigned*)&hbuf[m * kHID + k];
    }
    return au.h;
}

// ---------------------------------------------------------------------------
// Kernel 2: full encoder+decoder recurrence + fused vocab projection.
// grid = B/16 WGs, block = 128 threads (4 waves).
// ---------------------------------------------------------------------------
__global__ void __launch_bounds__(128)
seq2seq_kernel(const int*   __restrict__ s_tok,
               const int*   __restrict__ t_tok,
               const float* __restrict__ Genc,
               const float* __restrict__ Gdec,
               const float* __restrict__ encWhh,
               const float* __restrict__ decWhh,
               const float* __restrict__ fW,
               const float* __restrict__ fb,
               float*       __restrict__ out)
{
    extern __shared__ char smem[];
    int*       tokL = (int*)smem;                                   // 16*kS     ints
    float*     Gtab = (float*)(smem + 16 * kS * 4);                 // kVOCAB*kG4 floats
    float*     act  = (float*)(smem + 16 * kS * 4 + kVOCAB * kG4 * 4);   // 16*kG4 floats
    _Float16*  hbuf = (_Float16*)(smem + 16 * kS * 4 + kVOCAB * kG4 * 4
                                        + 16 * kG4 * 4);            // 16*kHID f16

    const int tid  = threadIdx.x;
    const int lane = tid & 31;
    const int wv   = tid >> 5;                 // wave id 0..3 -> gate i,f,g,o / vocab tile
    const int b0   = blockIdx.x * 16;          // batch tile base
    const int hi   = (lane >> 4) << 3;         // D-fragment row base for this lane half

    // Branchless per-wave activation constants:
    //   gates i,f,o (waves 0,1,3): sigmoid(x)  = 1*rcp(1+2^(-x*log2e)) + 0
    //   gate  g     (wave 2):      tanh(x)     = 2*rcp(1+2^(-2x*log2e)) - 1
    const bool  isTanh  = (wv == 2);
    const float actScl  = isTanh ? kN2L2E : kNL2E;   // multiplier inside exp2
    const float actAmp  = isTanh ? 2.0f   : 1.0f;
    const float actOff  = isTanh ? -1.0f  : 0.0f;

    // h0 = 0, c0 = 0
    for (int i = tid; i < 16 * kHID; i += 128) hbuf[i] = (_Float16)0.0f;
    float creg[8];
    #pragma unroll
    for (int q = 0; q < 8; ++q) creg[q] = 0.0f;

    // Constant projection fragments: B[k][n] = fW[n][k], n = vocab (padded to 64).
    const int vglob = wv * 16 + (lane & 15);
    const float fbv = (vglob < kVOCAB) ? fb[vglob] : 0.0f;
    v16h pb[2];
    #pragma unroll
    for (int kb = 0; kb < 2; ++kb) {
        #pragma unroll
        for (int e = 0; e < 16; ++e) {
            const int k = kb * 32 + ((lane >> 4) << 4) + e;
            pb[kb][e] = (vglob < kVOCAB) ? (_Float16)fW[vglob * kHID + k]
                                         : (_Float16)0.0f;
        }
    }

    for (int phase = 0; phase < 2; ++phase) {
        const int*   tok_g = phase ? t_tok  : s_tok;
        const float* Gsrc  = phase ? Gdec   : Genc;
        const float* Whh   = phase ? decWhh : encWhh;

        // Stage token tile [16][S] and gate LUT [VOCAB][4H] into LDS.
        for (int i = tid; i < 16 * kS; i += 128) {
            const int m = i >> 8, sx = i & (kS - 1);
            tokL[i] = tok_g[(b0 + m) * kS + sx];
        }
        for (int i = tid; i < kVOCAB * kG4; i += 128) Gtab[i] = Gsrc[i];

        // Recurrent-weight fragments for this wave's gate slice [64*wv, 64*wv+64):
        // B[k][n] = Whh[n][k]; pinned in VGPRs for all timesteps.
        v16h bf[4][2];
        #pragma unroll
        for (int nt = 0; nt < 4; ++nt) {
            const int n = wv * 64 + nt * 16 + (lane & 15);
            #pragma unroll
            for (int kb = 0; kb < 2; ++kb) {
                #pragma unroll
                for (int e = 0; e < 16; ++e) {
                    const int k = kb * 32 + ((lane >> 4) << 4) + e;
                    bf[nt][kb][e] = (_Float16)Whh[n * kHID + k];
                }
            }
        }
        __syncthreads();

        for (int sx = 0; sx < kS; ++sx) {
            // A fragments of h_{t-1}
            v16h a0 = load_hfrag(hbuf, 0, lane);
            v16h a1 = load_hfrag(hbuf, 1, lane);

            // tokens for the 8 batch rows this lane's D elements cover
            int tk[8];
            #pragma unroll
            for (int r = 0; r < 8; ++r) tk[r] = tokL[(hi + r) * kS + sx];

            // gate pre-activations: C init from LUT, then 2 chained WMMAs
            #pragma unroll
            for (int nt = 0; nt < 4; ++nt) {
                const int n = wv * 64 + nt * 16 + (lane & 15);
                v8f c;
                #pragma unroll
                for (int r = 0; r < 8; ++r) c[r] = Gtab[tk[r] * kG4 + n];
                c = __builtin_amdgcn_wmma_f32_16x16x32_f16(false, a0, false, bf[nt][0],
                                                           (short)0, c, false, false);
                c = __builtin_amdgcn_wmma_f32_16x16x32_f16(false, a1, false, bf[nt][1],
                                                           (short)0, c, false, false);
                #pragma unroll
                for (int r = 0; r < 8; ++r) {
                    // branchless sigmoid/tanh: amp * rcp(1 + 2^(x*scl)) + off
                    const float sg = fast_rcp(1.0f + fast_exp2(c[r] * actScl));
                    act[(hi + r) * kG4 + n] = __builtin_fmaf(actAmp, sg, actOff);
                }
            }
            __syncthreads();

            // LSTM cell update: 16*64 cells over 128 threads, c in registers
            #pragma unroll
            for (int q = 0; q < 8; ++q) {
                const int idx = tid * 8 + q;
                const int m = idx >> 6, j = idx & (kHID - 1);
                const float gi = act[m * kG4 + j];
                const float gf = act[m * kG4 + kHID + j];
                const float gg = act[m * kG4 + 2 * kHID + j];
                const float go = act[m * kG4 + 3 * kHID + j];
                creg[q] = __builtin_fmaf(gf, creg[q], gi * gg);
                const float hh = go * fast_tanh(creg[q]);
                hbuf[m * kHID + j] = (_Float16)hh;
            }
            __syncthreads();

            if (phase) {
                // fused projection: out[b, sx, :] = h_t @ fW^T + fb
                v16h pa0 = load_hfrag(hbuf, 0, lane);
                v16h pa1 = load_hfrag(hbuf, 1, lane);
                v8f d;
                #pragma unroll
                for (int r = 0; r < 8; ++r) d[r] = fbv;
                d = __builtin_amdgcn_wmma_f32_16x16x32_f16(false, pa0, false, pb[0],
                                                           (short)0, d, false, false);
                d = __builtin_amdgcn_wmma_f32_16x16x32_f16(false, pa1, false, pb[1],
                                                           (short)0, d, false, false);
                if (vglob < kVOCAB) {
                    #pragma unroll
                    for (int r = 0; r < 8; ++r) {
                        const int m = hi + r;
                        out[((size_t)(b0 + m) * kS + sx) * kVOCAB + vglob] = d[r];
                    }
                }
            }
        }
    }
}

// ---------------------------------------------------------------------------
// Host launcher
// ---------------------------------------------------------------------------
extern "C" void kernel_launch(void* const* d_in, const int* in_sizes, int n_in,
                              void* d_out, int out_size, void* d_ws, size_t ws_size,
                              hipStream_t stream) {
    const int*   s_tok = (const int*)d_in[0];
    const int*   t_tok = (const int*)d_in[1];
    const float* E     = (const float*)d_in[2];
    const float* eWih  = (const float*)d_in[3];
    const float* eWhh  = (const float*)d_in[4];
    const float* eBih  = (const float*)d_in[5];
    const float* eBhh  = (const float*)d_in[6];
    const float* dWih  = (const float*)d_in[7];
    const float* dWhh  = (const float*)d_in[8];
    const float* dBih  = (const float*)d_in[9];
    const float* dBhh  = (const float*)d_in[10];
    const float* fW    = (const float*)d_in[11];
    const float* fb    = (const float*)d_in[12];
    float* out = (float*)d_out;

    float* Genc = (float*)d_ws;                 // 50*256 floats
    float* Gdec = Genc + kVOCAB * kG4;          // 50*256 floats

    gate_table_kernel<<<1, kG4, 0, stream>>>(E, eWih, eBih, eBhh, Genc);
    gate_table_kernel<<<1, kG4, 0, stream>>>(E, dWih, dBih, dBhh, Gdec);

    const int smem_bytes = 16 * kS * 4            // token tile
                         + kVOCAB * kG4 * 4       // gate LUT
                         + 16 * kG4 * 4           // activation buffer
                         + 16 * kHID * 2;         // h buffer (f16)
    hipFuncSetAttribute((const void*)seq2seq_kernel,
                        hipFuncAttributeMaxDynamicSharedMemorySize, smem_bytes);

    seq2seq_kernel<<<kB / 16, 128, smem_bytes, stream>>>(
        s_tok, t_tok, Genc, Gdec, eWhh, dWhh, fW, fb, out);
}
